// GAT_22308060136200
// MI455X (gfx1250) — matrix operand order repaired and checked by reading, not compiled, not run on hardware
//
#include <hip/hip_runtime.h>
#include <hip/hip_fp16.h>

typedef float     v2f  __attribute__((ext_vector_type(2)));
typedef float     v4f  __attribute__((ext_vector_type(4)));
typedef float     v8f  __attribute__((ext_vector_type(8)));
typedef _Float16  v16h __attribute__((ext_vector_type(16)));

#define BATCH 256
#define NN    256
#define DIN   128
#define H     4
#define DH    32
#define HID   128   // H*DH
#define NEG_SLOPE 0.2f
#define LN_EPS    1e-5f

__device__ __forceinline__ float lrelu(float x) { return x > 0.f ? x : NEG_SLOPE * x; }

// --- CDNA5 async global->LDS copies (ASYNCcnt). Per ISA §10.2 aperture rules
// the low 32 bits of a generic LDS address are the LDS byte offset, so a
// truncated __shared__ pointer is a valid DS destination address.
__device__ __forceinline__ uint32_t lds_off(const void* p) {
    return (uint32_t)(uintptr_t)p;
}
__device__ __forceinline__ void async_b128(uint32_t dst_lds, const void* gptr) {
    asm volatile("global_load_async_to_lds_b128 %0, %1, off"
                 :: "v"(dst_lds), "v"((uint64_t)(uintptr_t)gptr) : "memory");
}
__device__ __forceinline__ void async_b32(uint32_t dst_lds, const void* gptr) {
    asm volatile("global_load_async_to_lds_b32 %0, %1, off"
                 :: "v"(dst_lds), "v"((uint64_t)(uintptr_t)gptr) : "memory");
}
__device__ __forceinline__ void wait_async0() {
    asm volatile("s_wait_asynccnt 0" ::: "memory");
}

// ---------------------------------------------------------------------------
// Kernel 1: h1[M x 128] = x[M x 128] @ W1[128 x 128], M = B*N.
// Exact f32 math via V_WMMA_F32_16X16X4_F32.
// Block = 256 threads (8 waves); each block computes a 128-row strip.
// K processed in 4 panels of 32; panels staged with async LDS loads.
// W panel stored TRANSPOSED (WsT[n][k]) so each B fragment is one ds_load_b64.
// A frag (16x4 f32): lane m = lane&15, VGPR v -> K = v + (lane>>4)*2.
// B frag (4x16 f32): lane n = lane&15, VGPR v -> K = v + (lane>>4)*2.
// ---------------------------------------------------------------------------
#define XS_STRIDE 36   // 32 + 4 pad (floats)
#define WT_STRIDE 36   // 32 + 4 pad (floats)

__global__ __launch_bounds__(256) void k1_gemm(const float* __restrict__ x,
                                               const float* __restrict__ W1,
                                               float* __restrict__ h1)
{
    __shared__ __align__(16) float Xs[128 * XS_STRIDE];   // [row][k]
    __shared__ __align__(16) float WsT[128 * WT_STRIDE];  // [n][k] (transposed)
    const int tid   = threadIdx.x;
    const int mBase = blockIdx.x * 128;
    const int wave  = tid >> 5;
    const int lane  = tid & 31;
    const int lm    = lane & 15;
    const int hi    = lane >> 4;
    const int mLoc  = wave * 16;

    v8f acc[8] = {};

    for (int kp = 0; kp < 4; ++kp) {
        // X panel: 128 rows x 32 K, 16B per lane, contiguous both sides
        #pragma unroll
        for (int t = 0; t < 4; ++t) {
            int i = tid + t * 256;                 // i in [0,1024)
            int row = i >> 3, c4 = (i & 7) << 2;
            async_b128(lds_off(&Xs[row * XS_STRIDE + c4]),
                       x + (size_t)(mBase + row) * DIN + kp * 32 + c4);
        }
        // W panel transposed: element (k,n) -> WsT[n][k]
        #pragma unroll
        for (int t = 0; t < 16; ++t) {
            int i = tid + t * 256;                 // i in [0,4096)
            int k = i >> 7, n = i & 127;           // coalesced read along n
            async_b32(lds_off(&WsT[n * WT_STRIDE + k]),
                      W1 + (size_t)(kp * 32 + k) * 128 + n);
        }
        wait_async0();
        __syncthreads();

        #pragma unroll
        for (int kk = 0; kk < 8; ++kk) {
            const int k = kk * 4 + hi * 2;
            const v2f a = *(const v2f*)(&Xs[(mLoc + lm) * XS_STRIDE + k]);
            #pragma unroll
            for (int nT = 0; nT < 8; ++nT) {
                const v2f b = *(const v2f*)(&WsT[(nT * 16 + lm) * WT_STRIDE + k]);
                acc[nT] = __builtin_amdgcn_wmma_f32_16x16x4_f32(
                    false, a, false, b, (short)0, acc[nT], false, false);
            }
        }
        __syncthreads();
    }

    // C/D layout: VGPR r -> row (r + hi*8), col = lane&15
    #pragma unroll
    for (int nT = 0; nT < 8; ++nT) {
        #pragma unroll
        for (int r = 0; r < 8; ++r) {
            int row = mBase + mLoc + r + hi * 8;
            h1[(size_t)row * HID + nT * 16 + lm] = acc[nT][r];
        }
    }
}

// ---------------------------------------------------------------------------
// Kernel 1b: per-node attention logits. a_src[n,h] = h1[n,h,:]·att_src[h,:]
// ---------------------------------------------------------------------------
__global__ __launch_bounds__(256) void k1b_attdots(const float* __restrict__ h1,
                                                   const float* __restrict__ att_src,
                                                   const float* __restrict__ att_dst,
                                                   float* __restrict__ asrc,
                                                   float* __restrict__ adst)
{
    __shared__ __align__(16) float as[HID];
    __shared__ __align__(16) float ad[HID];
    const int tid = threadIdx.x;
    if (tid < HID) { as[tid] = att_src[tid]; ad[tid] = att_dst[tid]; }
    __syncthreads();

    const int node = blockIdx.x * 256 + tid;
    const float* row = h1 + (size_t)node * HID;
    #pragma unroll
    for (int hh = 0; hh < H; ++hh) {
        float s = 0.f, d = 0.f;
        #pragma unroll
        for (int k = 0; k < DH; k += 4) {
            float4 v  = *(const float4*)(row + hh * DH + k);
            float4 a1 = *(const float4*)(&as[hh * DH + k]);
            float4 a2 = *(const float4*)(&ad[hh * DH + k]);
            s += v.x * a1.x + v.y * a1.y + v.z * a1.z + v.w * a1.w;
            d += v.x * a2.x + v.y * a2.y + v.z * a2.z + v.w * a2.w;
        }
        asrc[(size_t)node * H + hh] = s;
        adst[(size_t)node * H + hh] = d;
    }
}

// ---------------------------------------------------------------------------
// Kernel 2: fused softmax(lrelu(d_i+s_j)) @ h per (b,head) block.
// Flash-style: row max m_i = lrelu(d_i + max_j s_j) (lrelu is monotone).
// V staged transposed in LDS as f32 via async b32 copies; converted to f16
// at fragment-build time. P tiles built in-register as f16 A fragments;
// f32 accumulate via V_WMMA_F32_16X16X32_F16. Adds bias b1.
// A frag (16x32 f16): lane m = lane&15, elem e<8 -> K = e + hi*8,
//                     elem e>=8 -> K = (e-8) + 16 + hi*8.
// B frag (32x16 f16): lane n = lane&15, same K mapping as A frag.
// ---------------------------------------------------------------------------
#define HT_STRIDE 264   // 256 + 8 floats padding; keeps 16B alignment

__global__ __launch_bounds__(256) void k2_attn(const float* __restrict__ h1,
                                               const float* __restrict__ asrc,
                                               const float* __restrict__ adst,
                                               const float* __restrict__ b1,
                                               float* __restrict__ hagg)
{
    __shared__ __align__(16) float hT[DH * HT_STRIDE];  // [d][j], transposed f32
    __shared__ float sArr[NN];
    __shared__ float dArr[NN];
    __shared__ float red[8];

    const int tid  = threadIdx.x;
    const int b    = blockIdx.x >> 2;
    const int head = blockIdx.x & 3;
    const size_t base = (size_t)b * NN;

    // stage V transposed via async copies (coalesced reads along d)
    #pragma unroll
    for (int t = 0; t < 32; ++t) {
        int i = tid + t * 256;                 // i in [0, 8192)
        int d = i & (DH - 1);
        int j = i >> 5;
        async_b32(lds_off(&hT[d * HT_STRIDE + j]),
                  h1 + (base + j) * HID + head * DH + d);
    }
    sArr[tid] = asrc[(base + tid) * H + head];
    dArr[tid] = adst[(base + tid) * H + head];
    wait_async0();
    __syncthreads();

    // block max of s_j
    {
        float v = sArr[tid];
        #pragma unroll
        for (int off = 16; off; off >>= 1) v = fmaxf(v, __shfl_down(v, off, 32));
        if ((tid & 31) == 0) red[tid >> 5] = v;
        __syncthreads();
        if (tid == 0) {
            float m = red[0];
            #pragma unroll
            for (int w = 1; w < 8; ++w) m = fmaxf(m, red[w]);
            red[0] = m;
        }
        __syncthreads();
    }
    const float sm = red[0];

    const int wave = tid >> 5, lane = tid & 31, lm = lane & 15, hi = lane >> 4;
    const float bias0 = b1[head * DH + lm];
    const float bias1 = b1[head * DH + 16 + lm];

    #pragma unroll
    for (int it = 0; it < 2; ++it) {            // 2 row-tiles per wave
        const int iBase = (wave * 2 + it) * 16;
        const float di = dArr[iBase + lm];
        const float mi = lrelu(di + sm);
        float den = 0.f;
        v8f c0 = {}, c1 = {};

        for (int jb = 0; jb < NN; jb += 32) {   // K loop over neighbors
            v16h a;
            #pragma unroll
            for (int e = 0; e < 8; ++e) {
                float p = __expf(lrelu(di + sArr[jb + hi * 8 + e]) - mi);
                den += p;
                a[e] = (_Float16)p;
            }
            #pragma unroll
            for (int e = 8; e < 16; ++e) {
                float p = __expf(lrelu(di + sArr[jb + 16 + hi * 8 + (e - 8)]) - mi);
                den += p;
                a[e] = (_Float16)p;
            }
            // B fragments: 2 runs of 8 f32 per lane (K hi*8.., 16+hi*8..), convert to f16
            v16h b0, b1v;
            {
                const v4f* s0 = (const v4f*)(&hT[lm * HT_STRIDE + jb + hi * 8]);
                const v4f* s1 = (const v4f*)(&hT[(16 + lm) * HT_STRIDE + jb + hi * 8]);
                #pragma unroll
                for (int q = 0; q < 4; ++q) {
                    v4f f0 = s0[q + (q >> 1) * 2], f1 = s1[q + (q >> 1) * 2];
                    #pragma unroll
                    for (int c = 0; c < 4; ++c) {
                        b0[q * 4 + c]  = (_Float16)f0[c];
                        b1v[q * 4 + c] = (_Float16)f1[c];
                    }
                }
            }
            c0 = __builtin_amdgcn_wmma_f32_16x16x32_f16(false, a, false, b0,
                                                        (short)0, c0, false, false);
            c1 = __builtin_amdgcn_wmma_f32_16x16x32_f16(false, a, false, b1v,
                                                        (short)0, c1, false, false);
        }

        // row denominator is split across lane m (K 0-7,16-23 ...) and m+16
        #pragma unroll
        for (int r = 0; r < 8; ++r) {
            int rowm = r + hi * 8;
            float dfull = __shfl(den, rowm, 32) + __shfl(den, rowm + 16, 32);
            float inv = 1.f / dfull;
            size_t o = (base + iBase + rowm) * HID + head * DH;
            hagg[o + lm]      = c0[r] * inv + bias0;
            hagg[o + 16 + lm] = c1[r] * inv + bias1;
        }
    }
}

// ---------------------------------------------------------------------------
// Kernel 3: LayerNorm(HID) -> z = LN(h)·W2 (128->1) -> scalar GAT layer 2
// -> ELU. One block per batch, one thread per node.
// ---------------------------------------------------------------------------
__global__ __launch_bounds__(256) void k3_ln_gat2(const float* __restrict__ hagg,
                                                  const float* __restrict__ gamma,
                                                  const float* __restrict__ beta,
                                                  const float* __restrict__ W2,
                                                  const float* __restrict__ b2,
                                                  const float* __restrict__ as2,
                                                  const float* __restrict__ ad2,
                                                  float* __restrict__ out)
{
    __shared__ float g[HID], bt[HID], w2[HID];
    __shared__ float zsh[NN], ssh[NN], red[8];
    const int tid = threadIdx.x;
    const int b   = blockIdx.x;
    if (tid < HID) { g[tid] = gamma[tid]; bt[tid] = beta[tid]; w2[tid] = W2[tid]; }
    __syncthreads();

    const float* row = hagg + ((size_t)b * NN + tid) * HID;
    float s1 = 0.f, s2 = 0.f;
    #pragma unroll 4
    for (int k = 0; k < HID; k += 4) {
        float4 v = *(const float4*)(row + k);
        s1 += v.x + v.y + v.z + v.w;
        s2 += v.x * v.x + v.y * v.y + v.z * v.z + v.w * v.w;
    }
    const float mu   = s1 * (1.f / HID);
    const float var  = s2 * (1.f / HID) - mu * mu;
    const float rstd = rsqrtf(var + LN_EPS);
    float z = 0.f;
    #pragma unroll 4
    for (int k = 0; k < HID; k += 4) {
        float4 v = *(const float4*)(row + k);
        z += ((v.x - mu) * rstd * g[k]     + bt[k])     * w2[k];
        z += ((v.y - mu) * rstd * g[k + 1] + bt[k + 1]) * w2[k + 1];
        z += ((v.z - mu) * rstd * g[k + 2] + bt[k + 2]) * w2[k + 2];
        z += ((v.w - mu) * rstd * g[k + 3] + bt[k + 3]) * w2[k + 3];
    }
    zsh[tid] = z;
    ssh[tid] = z * as2[0];
    __syncthreads();

    // block max of s_j
    float v = ssh[tid];
    #pragma unroll
    for (int off = 16; off; off >>= 1) v = fmaxf(v, __shfl_down(v, off, 32));
    if ((tid & 31) == 0) red[tid >> 5] = v;
    __syncthreads();
    if (tid == 0) {
        float m = red[0];
        #pragma unroll
        for (int w = 1; w < 8; ++w) m = fmaxf(m, red[w]);
        red[0] = m;
    }
    __syncthreads();
    const float sm = red[0];

    const float di = z * ad2[0];
    const float mi = lrelu(di + sm);
    float acc = 0.f, den = 0.f;
    for (int j = 0; j < NN; ++j) {
        float w = __expf(lrelu(di + ssh[j]) - mi);
        acc += w * zsh[j];
        den += w;
    }
    float o = acc / den + b2[0];
    out[(size_t)b * NN + tid] = o > 0.f ? o : (__expf(o) - 1.f);
}

// ---------------------------------------------------------------------------
extern "C" void kernel_launch(void* const* d_in, const int* in_sizes, int n_in,
                              void* d_out, int out_size, void* d_ws, size_t ws_size,
                              hipStream_t stream)
{
    (void)in_sizes; (void)n_in; (void)out_size; (void)ws_size;
    const float* x        = (const float*)d_in[0];
    // d_in[1] = adj, unused (fully connected graph)
    const float* W1       = (const float*)d_in[2];
    const float* b1       = (const float*)d_in[3];
    const float* att_src1 = (const float*)d_in[4];
    const float* att_dst1 = (const float*)d_in[5];
    const float* gamma    = (const float*)d_in[6];
    const float* beta     = (const float*)d_in[7];
    const float* W2       = (const float*)d_in[8];
    const float* b2       = (const float*)d_in[9];
    const float* att_src2 = (const float*)d_in[10];
    const float* att_dst2 = (const float*)d_in[11];

    char* ws = (char*)d_ws;
    const size_t M = (size_t)BATCH * NN;
    float* h1   = (float*)ws;                                   // 33.5 MB
    float* hagg = (float*)(ws + M * HID * sizeof(float));       // 33.5 MB
    float* asrc = (float*)(ws + 2 * M * HID * sizeof(float));   // 1 MB
    float* adst = (float*)(ws + 2 * M * HID * sizeof(float) + M * H * sizeof(float));

    k1_gemm    <<<M / 128, 256, 0, stream>>>(x, W1, h1);
    k1b_attdots<<<M / 256, 256, 0, stream>>>(h1, att_src1, att_dst1, asrc, adst);
    k2_attn    <<<BATCH * H, 256, 0, stream>>>(h1, asrc, adst, b1, hagg);
    k3_ln_gat2 <<<BATCH, 256, 0, stream>>>(hagg, gamma, beta, W2, b2,
                                           att_src2, att_dst2, (float*)d_out);
}
